// GrabPhisher_36026185678968
// MI455X (gfx1250) — matrix operand order repaired and checked by reading, not compiled, not run on hardware
//
#include <hip/hip_runtime.h>
#include <hip/hip_bf16.h>
#include <float.h>

typedef __attribute__((ext_vector_type(16))) __bf16 v16bf;
typedef __attribute__((ext_vector_type(8)))  float  v8f;

#define HID 128

enum { EP_NONE = 0, EP_RELU = 1, EP_LN_RELU = 2, EP_RESID = 3 };

// Fused linear layer: C[M,128] = epilogue(A[M,K] @ W[128,K]^T + bias)
// Block = 256 threads (8 waves). Each block: 128 rows x 128 cols.
// Wave w owns rows [w*16, w*16+16) and all 8 column tiles of 16.
// Weights staged to LDS as bf16 in [k][n] layout; WMMA bf16, f32 accum.
__global__ __launch_bounds__(256) void gp_wmma_linear(
    const float* __restrict__ A, int lda,
    const float* __restrict__ W,           // [128, K] row-major
    const float* __restrict__ bias,        // [128]
    float* __restrict__ C, int ldc,
    int M, int K, int epilogue,
    const float* __restrict__ lng, const float* __restrict__ lnb,
    const float* __restrict__ resid, int ldr)
{
    extern __shared__ char gp_smem[];
    __bf16* sW = (__bf16*)gp_smem;         // [K][128]
    const int tid = threadIdx.x;
    // Stage W transposed into LDS as bf16: sW[k*128+n] = W[n*K+k]
    for (int idx = tid; idx < K * HID; idx += 256) {
        int n = idx / K;
        int k = idx - n * K;
        sW[k * HID + n] = (__bf16)W[idx];
    }
    __syncthreads();

    const int lane = tid & 31;
    const int wave = tid >> 5;
    const int half = lane >> 4;            // which 16-lane half
    const int l16  = lane & 15;
    const int rowBase = blockIdx.x * 128 + wave * 16;

    v8f acc[8];
    #pragma unroll
    for (int t = 0; t < 8; ++t)
        acc[t] = (v8f){0.f, 0.f, 0.f, 0.f, 0.f, 0.f, 0.f, 0.f};

    // A fragment rows: lane holds M = l16 (both halves), clamp for tail block
    int arow = rowBase + l16;
    if (arow > M - 1) arow = M - 1;
    const float* Arow = A + (size_t)arow * lda;

    for (int k0 = 0; k0 < K; k0 += 32) {
        // 16-bit A 16x32 layout: lanes 0-15 -> K = k0+{0..7, 16..23},
        //                        lanes 16-31 -> K = k0+{8..15, 24..31}
        const int kb = k0 + 8 * half;
        float4 a0 = *(const float4*)(Arow + kb);
        float4 a1 = *(const float4*)(Arow + kb + 4);
        float4 a2 = *(const float4*)(Arow + kb + 16);
        float4 a3 = *(const float4*)(Arow + kb + 20);
        v16bf af;
        af[0]  = (__bf16)a0.x; af[1]  = (__bf16)a0.y; af[2]  = (__bf16)a0.z; af[3]  = (__bf16)a0.w;
        af[4]  = (__bf16)a1.x; af[5]  = (__bf16)a1.y; af[6]  = (__bf16)a1.z; af[7]  = (__bf16)a1.w;
        af[8]  = (__bf16)a2.x; af[9]  = (__bf16)a2.y; af[10] = (__bf16)a2.z; af[11] = (__bf16)a2.w;
        af[12] = (__bf16)a3.x; af[13] = (__bf16)a3.y; af[14] = (__bf16)a3.z; af[15] = (__bf16)a3.w;

        // B fragment: lane N = t*16+l16, K = k0+{0..7,16..23} (lanes 0-15) / k0+{8..15,24..31} (lanes 16-31)
        const __bf16* sr = sW + (size_t)(k0 + 8 * half) * HID + l16;
        #pragma unroll
        for (int t = 0; t < 8; ++t) {
            v16bf bf;
            #pragma unroll
            for (int s = 0; s < 16; ++s) bf[s] = sr[(size_t)(s < 8 ? s : s + 8) * HID + t * 16];
            acc[t] = __builtin_amdgcn_wmma_f32_16x16x32_bf16(
                false, af, false, bf, (short)0, acc[t], false, false);
        }
    }

    float bv[8];
    #pragma unroll
    for (int t = 0; t < 8; ++t) bv[t] = bias[t * 16 + l16];

    if (epilogue == EP_LN_RELU) {
        // Each wave holds full rows: VGPR r, lanes 0-15 -> row r; lanes 16-31 -> row r+8.
        // Reduce 8 tile values per lane + 16 lanes via xor shuffles (stay in half).
        float gv[8], ov[8];
        #pragma unroll
        for (int t = 0; t < 8; ++t) { gv[t] = lng[t * 16 + l16]; ov[t] = lnb[t * 16 + l16]; }
        #pragma unroll
        for (int r = 0; r < 8; ++r) {
            float s = 0.f;
            #pragma unroll
            for (int t = 0; t < 8; ++t) { float v = acc[t][r] + bv[t]; acc[t][r] = v; s += v; }
            s += __shfl_xor(s, 1, 32); s += __shfl_xor(s, 2, 32);
            s += __shfl_xor(s, 4, 32); s += __shfl_xor(s, 8, 32);
            float mu = s * (1.f / 128.f);
            float s2 = 0.f;
            #pragma unroll
            for (int t = 0; t < 8; ++t) { float d = acc[t][r] - mu; s2 += d * d; }
            s2 += __shfl_xor(s2, 1, 32); s2 += __shfl_xor(s2, 2, 32);
            s2 += __shfl_xor(s2, 4, 32); s2 += __shfl_xor(s2, 8, 32);
            float inv = rsqrtf(s2 * (1.f / 128.f) + 1e-5f);
            #pragma unroll
            for (int t = 0; t < 8; ++t) {
                float v = (acc[t][r] - mu) * inv * gv[t] + ov[t];
                acc[t][r] = v > 0.f ? v : 0.f;
            }
        }
    } else {
        #pragma unroll
        for (int r = 0; r < 8; ++r) {
            const int row = rowBase + r + 8 * half;
            #pragma unroll
            for (int t = 0; t < 8; ++t) {
                float v = acc[t][r] + bv[t];
                if (epilogue == EP_RELU) v = v > 0.f ? v : 0.f;
                if (epilogue == EP_RESID && row < M)
                    v += resid[(size_t)row * ldr + t * 16 + l16];
                acc[t][r] = v;
            }
        }
    }

    #pragma unroll
    for (int r = 0; r < 8; ++r) {
        const int row = rowBase + r + 8 * half;
        if (row < M) {
            float* Crow = C + (size_t)row * ldc;
            #pragma unroll
            for (int t = 0; t < 8; ++t) Crow[t * 16 + l16] = acc[t][r];
        }
    }
}

// Zero the neighbor half (cols 128..255) of the stride-256 h buffer.
__global__ __launch_bounds__(256) void gp_zero_nbr(float* __restrict__ hbuf, int Nn)
{
    long t = (long)blockIdx.x * blockDim.x + threadIdx.x;
    long row = t >> 5;
    if (row >= Nn) return;
    int c = (int)(t & 31) * 4;
    *(float4*)(hbuf + row * 256 + 128 + c) = make_float4(0.f, 0.f, 0.f, 0.f);
}

// segment_sum(h[row], col): 32 lanes per edge, float4 per lane, atomic adds
// into cols 128..255 of the stride-256 h buffer (building hcat implicitly).
__global__ __launch_bounds__(256) void gp_scatter_edges(
    const int* __restrict__ ei, int E, float* __restrict__ hbuf)
{
    long t = (long)blockIdx.x * blockDim.x + threadIdx.x;
    long e = t >> 5;
    if (e >= E) return;
    int c = (int)(t & 31) * 4;
    int r = ei[e];
    int d = ei[(long)E + e];
    float4 v = *(const float4*)(hbuf + (long)r * 256 + c);
    float* dst = hbuf + (long)d * 256 + 128 + c;
    atomicAdd(dst + 0, v.x);
    atomicAdd(dst + 1, v.y);
    atomicAdd(dst + 2, v.z);
    atomicAdd(dst + 3, v.w);
}

// Pooling stage 1: per-block partial sum + max over rows, per channel.
__global__ __launch_bounds__(128) void gp_pool1(
    const float* __restrict__ hbuf, int Nn, float* __restrict__ part)
{
    int c = threadIdx.x;                    // 0..127
    float s = 0.f, mx = -FLT_MAX;
    for (long r = blockIdx.x; r < Nn; r += gridDim.x) {
        float v = hbuf[r * 256 + c];
        s += v;
        mx = fmaxf(mx, v);
    }
    part[(size_t)blockIdx.x * 256 + c] = s;
    part[(size_t)blockIdx.x * 256 + 128 + c] = mx;
}

// Pooling stage 2 + classifier, single block.
__global__ __launch_bounds__(256) void gp_pool2_classify(
    const float* __restrict__ part, int nblocks, int Nn,
    const float* __restrict__ c_w1, const float* __restrict__ c_b1,
    const float* __restrict__ c_w2, const float* __restrict__ c_b2,
    float* __restrict__ out)
{
    __shared__ float g[256];
    __shared__ float hc[128];
    int t = threadIdx.x;
    if (t < 128) {
        float s = 0.f;
        for (int b = 0; b < nblocks; ++b) s += part[(size_t)b * 256 + t];
        g[t] = s / (float)Nn;               // mean pool
    } else {
        int c = t - 128;
        float mx = -FLT_MAX;
        for (int b = 0; b < nblocks; ++b) mx = fmaxf(mx, part[(size_t)b * 256 + 128 + c]);
        g[t] = mx;                          // max pool
    }
    __syncthreads();
    if (t < 128) {
        float s = c_b1[t];
        const float* wr = c_w1 + (size_t)t * 256;
        for (int k = 0; k < 256; ++k) s += g[k] * wr[k];
        hc[t] = s > 0.f ? s : 0.f;
    }
    __syncthreads();
    if (t < 2) {
        float s = c_b2[t];
        const float* wr = c_w2 + (size_t)t * 128;
        for (int j = 0; j < 128; ++j) s += hc[j] * wr[j];
        out[t] = s;
    }
}

extern "C" void kernel_launch(void* const* d_in, const int* in_sizes, int n_in,
                              void* d_out, int out_size, void* d_ws, size_t ws_size,
                              hipStream_t stream) {
    const float* x        = (const float*)d_in[0];
    const int*   ei       = (const int*)d_in[1];
    // d_in[2] = batch (all zeros, single graph) -> unused
    const float* fe_w1    = (const float*)d_in[3];
    const float* fe_b1    = (const float*)d_in[4];
    const float* fe_w2    = (const float*)d_in[5];
    const float* fe_b2    = (const float*)d_in[6];
    const float* conv_w   = (const float*)d_in[7];
    const float* conv_b   = (const float*)d_in[8];
    const float* ln_g     = (const float*)d_in[9];
    const float* ln_b     = (const float*)d_in[10];
    const float* in_proj_w= (const float*)d_in[11];
    const float* in_proj_b= (const float*)d_in[12];
    const float* out_w    = (const float*)d_in[13];
    const float* out_b    = (const float*)d_in[14];
    const float* c_w1     = (const float*)d_in[15];
    const float* c_b1     = (const float*)d_in[16];
    const float* c_w2     = (const float*)d_in[17];
    const float* c_b2     = (const float*)d_in[18];

    const int Nn = in_sizes[0] / 256;       // N nodes (IN = 256)
    const int E  = in_sizes[1] / 2;

    float* hbuf = (float*)d_ws;                       // [Nn, 256]: h | nbr
    float* bufA = hbuf + (size_t)Nn * 256;            // [Nn, 128] temp
    float* part = bufA + (size_t)Nn * 128;            // [512, 256] pool partials

    const int gM = (Nn + 127) / 128;
    const size_t lds256 = 256 * HID * sizeof(__bf16); // 64 KB
    const size_t lds128 = 128 * HID * sizeof(__bf16); // 32 KB

    // feature extractor: t0 = relu(x @ fe_w1^T + b1); h = t0 @ fe_w2^T + b2
    gp_wmma_linear<<<gM, 256, lds256, stream>>>(x, 256, fe_w1, fe_b1, bufA, 128,
        Nn, 256, EP_RELU, nullptr, nullptr, nullptr, 0);
    gp_wmma_linear<<<gM, 256, lds128, stream>>>(bufA, 128, fe_w2, fe_b2, hbuf, 256,
        Nn, 128, EP_NONE, nullptr, nullptr, nullptr, 0);

    // 3 x graph conv: nbr scatter-add into cols 128..255, then fused GEMM+LN+ReLU
    const long eThreads = (long)E * 32;
    const int  eBlocks  = (int)((eThreads + 255) / 256);
    const int  zBlocks  = (int)(((long)Nn * 32 + 255) / 256);
    for (int i = 0; i < 3; ++i) {
        gp_zero_nbr<<<zBlocks, 256, 0, stream>>>(hbuf, Nn);
        gp_scatter_edges<<<eBlocks, 256, 0, stream>>>(ei, E, hbuf);
        gp_wmma_linear<<<gM, 256, lds256, stream>>>(hbuf, 256,
            conv_w + (size_t)i * HID * 256, conv_b + (size_t)i * HID, hbuf, 256,
            Nn, 256, EP_LN_RELU, ln_g + (size_t)i * HID, ln_b + (size_t)i * HID,
            nullptr, 0);
    }

    // Attention with seq_len 1: softmax == 1 -> attn = v; q,k are dead code.
    // v = h @ Wv^T + bv (Wv = in_proj_w rows 256..383); h += v @ out_w^T + out_b
    gp_wmma_linear<<<gM, 256, lds128, stream>>>(hbuf, 256,
        in_proj_w + (size_t)256 * HID, in_proj_b + 256, bufA, 128,
        Nn, 128, EP_NONE, nullptr, nullptr, nullptr, 0);
    gp_wmma_linear<<<gM, 256, lds128, stream>>>(bufA, 128, out_w, out_b, hbuf, 256,
        Nn, 128, EP_RESID, nullptr, nullptr, hbuf, 256);

    // mean/max pool + classifier
    gp_pool1<<<512, 128, 0, stream>>>(hbuf, Nn, part);
    gp_pool2_classify<<<1, 256, 0, stream>>>(part, 512, Nn,
        c_w1, c_b1, c_w2, c_b2, (float*)d_out);
}
